// FP8Linear_1975684956485
// MI455X (gfx1250) — compile-verified, matrix operand order checked by reading, not run on hardware
//
#include <hip/hip_runtime.h>

// ---------------------------------------------------------------------------
// FP8 (e4m3) linear:  out[M,N] = q(x)[M,K] @ q(W)[N,K]^T + bias,  fp32 out
//   M = 8*4096 = 32768, N = 1024, K = 1024
// Pass 1: fp32 -> packed e4m3 (v_cvt_pk_fp8_f32), clamp +-448 per reference.
// Pass 2: double-buffered LDS GEMM on v_wmma_f32_16x16x128_fp8_fp8 with
//         GLOBAL_LOAD_ASYNC_TO_LDS_B128 staging (ASYNCcnt) when available.
// ---------------------------------------------------------------------------

typedef __attribute__((ext_vector_type(16))) int   v16i;
typedef __attribute__((ext_vector_type(8)))  float v8f;

#define GEMM_M 32768
#define GEMM_N 1024
#define GEMM_K 1024
#define BM 128
#define BN 128
#define BK 128
#define KSTEPS (GEMM_K / BK)       // 8
#define LDS_STRIDE 144             // 128 + 16B pad: fragment rows hit distinct banks

#if defined(__gfx1250__) && __has_builtin(__builtin_amdgcn_global_load_async_to_lds_b128)
#define USE_ASYNC_LDS 1
#else
#define USE_ASYNC_LDS 0
#endif

#if USE_ASYNC_LDS
// Builtin signature (from hipcc diagnostic / upstream tests):
//   void __builtin_amdgcn_global_load_async_to_lds_b128(
//       v4i __attribute__((address_space(1)))* src,   // global
//       v4i __attribute__((address_space(3)))* dst,   // LDS
//       Imm int offset, Imm int cpol);
typedef int v4i_gcc __attribute__((vector_size(16)));
typedef v4i_gcc __attribute__((address_space(1)))* g_v4_ptr;
typedef v4i_gcc __attribute__((address_space(3)))* l_v4_ptr;
#define ASYNC_CP16(g, l, off)                                                  \
  __builtin_amdgcn_global_load_async_to_lds_b128(                              \
      (g_v4_ptr)(g), (l_v4_ptr)(l), off, 0)
#endif

// ---------------- fp32 -> e4m3 quantization ----------------

__device__ __forceinline__ unsigned int f32_to_e4m3_manual(float f) {
  unsigned int u  = __float_as_uint(f);
  unsigned int s  = (u >> 24) & 0x80u;
  unsigned int au = u & 0x7FFFFFFFu;
  if (au == 0u) return s;
  int exp = (int)(au >> 23) - 127;
  unsigned int mant = au & 0x7FFFFFu;
  unsigned int out;
  if (exp < -9) {
    out = 0u;
  } else if (exp < -6) {                        // subnormal e4m3
    int shift = (20 - 6) - exp;                 // 21..23
    unsigned int m    = mant | 0x800000u;
    unsigned int q    = m >> shift;
    unsigned int rem  = m & ((1u << shift) - 1u);
    unsigned int h    = 1u << (shift - 1);
    if (rem > h || (rem == h && (q & 1u))) q++;
    out = q;
  } else {
    unsigned int q   = mant >> 20;
    unsigned int rem = mant & 0xFFFFFu;
    if (rem > 0x80000u || (rem == 0x80000u && (q & 1u))) q++;
    unsigned int e = (unsigned int)(exp + 7);
    if (q == 8u) { q = 0u; e++; }
    if (e > 15u || (e == 15u && q > 6u)) { e = 15u; q = 6u; }  // clamp 448
    out = (e << 3) | q;
  }
  return out | s;
}

__device__ __forceinline__ unsigned int pack4_e4m3(float4 f) {
  f.x = fminf(fmaxf(f.x, -448.f), 448.f);
  f.y = fminf(fmaxf(f.y, -448.f), 448.f);
  f.z = fminf(fmaxf(f.z, -448.f), 448.f);
  f.w = fminf(fmaxf(f.w, -448.f), 448.f);
#if __has_builtin(__builtin_amdgcn_cvt_pk_fp8_f32)
  unsigned int w = (unsigned int)__builtin_amdgcn_cvt_pk_fp8_f32(f.x, f.y, 0, false);
  w = (unsigned int)__builtin_amdgcn_cvt_pk_fp8_f32(f.z, f.w, (int)w, true);
  return w;
#else
  return  f32_to_e4m3_manual(f.x)
       | (f32_to_e4m3_manual(f.y) << 8)
       | (f32_to_e4m3_manual(f.z) << 16)
       | (f32_to_e4m3_manual(f.w) << 24);
#endif
}

__global__ void __launch_bounds__(256)
quant_fp8_kernel(const float* __restrict__ in, unsigned int* __restrict__ out,
                 int nwords) {
  int i = blockIdx.x * blockDim.x + threadIdx.x;
  if (i >= nwords) return;
  float4 f = reinterpret_cast<const float4*>(in)[i];   // global_load_b128
  out[i] = pack4_e4m3(f);
}

// ---------------- staging helpers ----------------

// Copy one 64B half-row chunk of A and of W into the LDS tile buffers.
// Async path: 8 GLOBAL_LOAD_ASYNC_TO_LDS_B128 per thread (ASYNCcnt batch of 8).
__device__ __forceinline__ void stage_tile(const unsigned char* gA,
                                           const unsigned char* gW,
                                           unsigned char* lA,
                                           unsigned char* lW) {
#if USE_ASYNC_LDS
  ASYNC_CP16(gA, lA, 0);
  ASYNC_CP16(gA, lA, 16);
  ASYNC_CP16(gA, lA, 32);
  ASYNC_CP16(gA, lA, 48);
  ASYNC_CP16(gW, lW, 0);
  ASYNC_CP16(gW, lW, 16);
  ASYNC_CP16(gW, lW, 32);
  ASYNC_CP16(gW, lW, 48);
#else
#pragma unroll
  for (int q = 0; q < 4; ++q) {
    uint4 va = reinterpret_cast<const uint4*>(gA)[q];
    reinterpret_cast<uint4*>(lA)[q] = va;
    uint4 vb = reinterpret_cast<const uint4*>(gW)[q];
    reinterpret_cast<uint4*>(lW)[q] = vb;
  }
#endif
}

__device__ __forceinline__ void wait_async_le8() {
#if USE_ASYNC_LDS
#if __has_builtin(__builtin_amdgcn_s_wait_asynccnt)
  __builtin_amdgcn_s_wait_asynccnt(8);
#else
  asm volatile("s_wait_asynccnt 0x8" ::: "memory");
#endif
#endif
}

__device__ __forceinline__ void wait_async_le0() {
#if USE_ASYNC_LDS
#if __has_builtin(__builtin_amdgcn_s_wait_asynccnt)
  __builtin_amdgcn_s_wait_asynccnt(0);
#else
  asm volatile("s_wait_asynccnt 0x0" ::: "memory");
#endif
#endif
}

// ---------------- fp8 WMMA GEMM ----------------

__global__ void __launch_bounds__(256)
fp8_gemm_wmma(const unsigned char* __restrict__ A,   // [M,K] e4m3
              const unsigned char* __restrict__ W,   // [N,K] e4m3
              const float* __restrict__ bias,        // [N]
              float* __restrict__ out) {             // [M,N] f32
  __shared__ unsigned char As[2][BM * LDS_STRIDE];   // 2 x 18 KB
  __shared__ unsigned char Bs[2][BN * LDS_STRIDE];   // 2 x 18 KB

  const int tid   = threadIdx.x;
  const int lane  = tid & 31;
  const int wave  = tid >> 5;
  const int waveM = wave & 3;        // 4 waves along M: 32-row slabs
  const int waveN = wave >> 2;       // 2 waves along N: 64-col slabs
  const int half  = lane >> 4;       // K half (0: K0-63, 1: K64-127)
  const int l16   = lane & 15;

  const int    blockN0 = blockIdx.x * BN;
  const size_t blockM0 = (size_t)blockIdx.y * BM;

  // loader mapping: thread -> (row, 64B half-row chunk)
  const int ldRow = tid >> 1;
  const int ldCol = (tid & 1) * 64;

  const unsigned char* aSrc = A + (blockM0 + (size_t)ldRow) * GEMM_K + ldCol;
  const unsigned char* wSrc = W + ((size_t)blockN0 + ldRow) * GEMM_K + ldCol;
  const int ldsOff = ldRow * LDS_STRIDE + ldCol;

  v8f acc[2][4];
  const v8f vzero = {0.f, 0.f, 0.f, 0.f, 0.f, 0.f, 0.f, 0.f};
#pragma unroll
  for (int i = 0; i < 2; ++i)
#pragma unroll
    for (int j = 0; j < 4; ++j) acc[i][j] = vzero;

  union Frag { v16i v; uint4 q[4]; };

  // Prologue: stage K-step 0 into buffer 0.
  stage_tile(aSrc, wSrc, &As[0][ldsOff], &Bs[0][ldsOff]);

#pragma unroll
  for (int kt = 0; kt < KSTEPS; ++kt) {
    const int cur = kt & 1;
    if (kt + 1 < KSTEPS) {
      // Stage next K-step into the other buffer, then wait for the *current*
      // batch only (<= 8 outstanding leaves the new batch in flight).
      stage_tile(aSrc + (kt + 1) * BK, wSrc + (kt + 1) * BK,
                 &As[cur ^ 1][ldsOff], &Bs[cur ^ 1][ldsOff]);
      wait_async_le8();
    } else {
      wait_async_le0();
    }
    __syncthreads();   // all waves' portions of the current tile are in LDS

    Frag fa[2], fb[4];
#pragma unroll
    for (int i = 0; i < 2; ++i) {      // A: 16x128 fragment per M-tile
      const uint4* p = reinterpret_cast<const uint4*>(
          &As[cur][(waveM * 32 + i * 16 + l16) * LDS_STRIDE + half * 64]);
#pragma unroll
      for (int q = 0; q < 4; ++q) fa[i].q[q] = p[q];
    }
#pragma unroll
    for (int j = 0; j < 4; ++j) {      // B: 128x16 fragment per N-tile
      const uint4* p = reinterpret_cast<const uint4*>(
          &Bs[cur][(waveN * 64 + j * 16 + l16) * LDS_STRIDE + half * 64]);
#pragma unroll
      for (int q = 0; q < 4; ++q) fb[j].q[q] = p[q];
    }

#pragma unroll
    for (int i = 0; i < 2; ++i)
#pragma unroll
      for (int j = 0; j < 4; ++j)
        acc[i][j] = __builtin_amdgcn_wmma_f32_16x16x128_fp8_fp8(
            fa[i].v, fb[j].v, (short)0, acc[i][j], false, false);

    __syncthreads();   // everyone done reading `cur` before it is overwritten
  }

  // epilogue: C layout -> M = 8*half + r, N = l16
#pragma unroll
  for (int j = 0; j < 4; ++j) {
    int n = blockN0 + waveN * 64 + j * 16 + l16;
    float bv = bias[n];
#pragma unroll
    for (int i = 0; i < 2; ++i) {
      size_t mBase = blockM0 + (size_t)(waveM * 32 + i * 16 + half * 8);
#pragma unroll
      for (int r = 0; r < 8; ++r)
        out[(mBase + r) * GEMM_N + n] = acc[i][j][r] + bv;
    }
  }
}

// ---------------- launch ----------------

extern "C" void kernel_launch(void* const* d_in, const int* in_sizes, int n_in,
                              void* d_out, int out_size, void* d_ws, size_t ws_size,
                              hipStream_t stream) {
  const float* x    = (const float*)d_in[0];   // (8,4096,1024) f32
  const float* w    = (const float*)d_in[1];   // (1024,1024)  f32
  const float* bias = (const float*)d_in[2];   // (1024,)      f32
  float*       out  = (float*)d_out;           // (8,4096,1024) f32

  unsigned char* a8 = (unsigned char*)d_ws;                  // 32 MB
  unsigned char* w8 = a8 + (size_t)GEMM_M * GEMM_K;          // +1 MB

  const int xWords = GEMM_M * GEMM_K / 4;   // 8,388,608 packed dwords
  const int wWords = GEMM_N * GEMM_K / 4;   //   262,144
  quant_fp8_kernel<<<(xWords + 255) / 256, 256, 0, stream>>>(
      x, (unsigned int*)a8, xWords);
  quant_fp8_kernel<<<(wWords + 255) / 256, 256, 0, stream>>>(
      w, (unsigned int*)w8, wWords);

  dim3 grid(GEMM_N / BN, GEMM_M / BM);      // (8, 256)
  fp8_gemm_wmma<<<grid, 256, 0, stream>>>(a8, w8, bias, out);
}
